// Qwen3MoeAttention_39256001086125
// MI455X (gfx1250) — matrix operand order repaired and checked by reading, not compiled, not run on hardware
//
#include <hip/hip_runtime.h>
#include <hip/hip_bf16.h>

#define T_     2048
#define H_     32
#define HKV_   4
#define D_     128
#define HID_   2048
#define QKVW_  5120   // (H + 2*HKV) * D
#define NOUT_  4096   // H * D

typedef __attribute__((ext_vector_type(16))) __bf16 bf16x16;
typedef __attribute__((ext_vector_type(8)))  __bf16 bf16x8;
typedef __attribute__((ext_vector_type(8)))  float  f32x8;
typedef __attribute__((ext_vector_type(4)))  float  f32x4;

__device__ static inline __bf16 f2bf(float f) {
    unsigned u = __builtin_bit_cast(unsigned, f);
    u += 0x7fffu + ((u >> 16) & 1u);           // round-to-nearest-even
    unsigned short s = (unsigned short)(u >> 16);
    return __builtin_bit_cast(__bf16, s);
}

__device__ static inline f32x8 wmma_bf16(bf16x16 a, bf16x16 b, f32x8 c) {
    return __builtin_amdgcn_wmma_f32_16x16x32_bf16(
        false, a, false, b, (short)0, c, false, false);
}

// A-matrix fragment (16x32 bf16): lanes 0-15 row=lane, K chunks {0..7,16..23};
// lanes 16-31 row=lane-16, K chunks {8..15,24..31}.
__device__ static inline bf16x16 ldfragA(const __bf16* base, int stride, int koff, int lane) {
    int r  = lane & 15;
    int k0 = koff + ((lane >> 4) << 3);
    union { bf16x16 v; bf16x8 h[2]; } u;
    u.h[0] = *(const bf16x8*)(base + r * stride + k0);
    u.h[1] = *(const bf16x8*)(base + r * stride + k0 + 16);
    return u.v;
}

// B-matrix fragment (32x16 bf16) read from LDS tile stored transposed [n][k]:
// lanes 0-15 col=lane hold K chunks {0..7,16..23}; lanes 16-31 hold K chunks {8..15,24..31}.
__device__ static inline bf16x16 ldfragB(const __bf16* base, int stride, int koff, int lane) {
    int n  = lane & 15;
    int k0 = koff + ((lane >> 4) << 3);
    union { bf16x16 v; bf16x8 h[2]; } u;
    u.h[0] = *(const bf16x8*)(base + n * stride + k0);
    u.h[1] = *(const bf16x8*)(base + n * stride + k0 + 16);
    return u.v;
}

// Async direct global->LDS 16-byte copy (CDNA5 ASYNCcnt path, GV mode).
// LDS byte address = flat pointer truncated to 32 bits (ISA aperture rule).
__device__ static inline void async_g2l_b128(const void* gptr, void* lptr) {
    unsigned           lds = (unsigned)(unsigned long long)lptr;
    unsigned long long ga  = (unsigned long long)gptr;
    asm volatile("global_load_async_to_lds_b128 %0, %1, off"
                 :: "v"(lds), "v"(ga) : "memory");
}

// ---------------------------------------------------------------------------
// Kernel 1/4: C[M,N] = A[M,K] * B[K,N], A fp32-or-bf16, B fp32, C fp32.
// Block 256 thr (8 waves), tile 128x128x32, waves 2x4, each wave 64x32.
// (A/B need fp32->bf16 conversion in flight, so staging goes through VGPRs.)
// ---------------------------------------------------------------------------
#define BM 128
#define BN 128
#define BK 32
#define AST 48   // LDS row stride (bf16 elems): 96B, 16B-aligned chunks
#define BST 48

template <bool ABF16>
__global__ __launch_bounds__(256)
void gemm_wmma(const void* __restrict__ Ap, const float* __restrict__ B,
               float* __restrict__ C, int M, int N, int K) {
    __shared__ __bf16 As[BM * AST];
    __shared__ __bf16 Bs[BN * BST];   // stored transposed: [n][k]

    const int tid  = threadIdx.x;
    const int lane = tid & 31;
    const int wv   = tid >> 5;
    const int wr   = wv >> 2;   // 0..1
    const int wc   = wv & 3;    // 0..3
    const int row0 = blockIdx.y * BM;
    const int col0 = blockIdx.x * BN;

    const float*  Af = (const float*)Ap;
    const __bf16* Ab = (const __bf16*)Ap;

    f32x8 acc[4][2];
#pragma unroll
    for (int mi = 0; mi < 4; ++mi)
#pragma unroll
        for (int ni = 0; ni < 2; ++ni)
#pragma unroll
            for (int i = 0; i < 8; ++i) acc[mi][ni][i] = 0.0f;

    for (int kb = 0; kb < K; kb += BK) {
        // ---- stage A tile (128x32) as bf16 ----
#pragma unroll
        for (int it = 0; it < 4; ++it) {
            int slot = tid + it * 256;
            int r = slot >> 3;
            int c = (slot & 7) << 2;
            __bf16* dst = &As[r * AST + c];
            if (ABF16) {
                const __bf16* src = Ab + (size_t)(row0 + r) * K + kb + c;
                dst[0] = src[0]; dst[1] = src[1]; dst[2] = src[2]; dst[3] = src[3];
            } else {
                f32x4 v = *(const f32x4*)(Af + (size_t)(row0 + r) * K + kb + c);
                dst[0] = f2bf(v.x); dst[1] = f2bf(v.y);
                dst[2] = f2bf(v.z); dst[3] = f2bf(v.w);
            }
        }
        // ---- stage B tile (32x128) transposed into [n][k] bf16 ----
#pragma unroll
        for (int it = 0; it < 4; ++it) {
            int slot = tid + it * 256;
            int r = slot >> 5;            // k row 0..31
            int c = (slot & 31) << 2;     // n 0..124
            f32x4 v = *(const f32x4*)(B + (size_t)(kb + r) * N + col0 + c);
            Bs[(c + 0) * BST + r] = f2bf(v.x);
            Bs[(c + 1) * BST + r] = f2bf(v.y);
            Bs[(c + 2) * BST + r] = f2bf(v.z);
            Bs[(c + 3) * BST + r] = f2bf(v.w);
        }
        if (kb + BK < K) {  // hint next tiles into cache (global_prefetch_b8)
            __builtin_prefetch(Af + (size_t)(row0 + (tid >> 3)) * K + kb + BK, 0, 1);
            __builtin_prefetch(B + (size_t)(kb + BK + (tid >> 5)) * N + col0 + ((tid & 31) << 2), 0, 1);
        }
        __syncthreads();

        bf16x16 afr[4], bfr[2];
#pragma unroll
        for (int mi = 0; mi < 4; ++mi)
            afr[mi] = ldfragA(&As[(wr * 64 + mi * 16) * AST], AST, 0, lane);
#pragma unroll
        for (int ni = 0; ni < 2; ++ni)
            bfr[ni] = ldfragB(&Bs[(wc * 32 + ni * 16) * BST], BST, 0, lane);
#pragma unroll
        for (int mi = 0; mi < 4; ++mi)
#pragma unroll
            for (int ni = 0; ni < 2; ++ni)
                acc[mi][ni] = wmma_bf16(afr[mi], bfr[ni], acc[mi][ni]);
        __syncthreads();
    }

    const int g   = lane >> 4;
    const int nlo = lane & 15;
#pragma unroll
    for (int mi = 0; mi < 4; ++mi)
#pragma unroll
        for (int ni = 0; ni < 2; ++ni)
#pragma unroll
            for (int i = 0; i < 8; ++i) {
                int row = row0 + wr * 64 + mi * 16 + i + 8 * g;
                int col = col0 + wc * 32 + ni * 16 + nlo;
                C[(size_t)row * N + col] = acc[mi][ni][i];
            }
}

// ---------------------------------------------------------------------------
// Kernel 2/4: RMSNorm + RoPE on Q/K, transpose V; fp32 qkv -> bf16 layouts.
// Block (64,4): 64 lanes per row-task (each lane owns d=j and d=j+64).
// ---------------------------------------------------------------------------
__global__ __launch_bounds__(256)
void qkv_post(const float* __restrict__ qkv, const int* __restrict__ pos,
              const float* __restrict__ qw, const float* __restrict__ kw,
              __bf16* __restrict__ qB, __bf16* __restrict__ kB,
              __bf16* __restrict__ vT) {
    __shared__ float red[4][2];
    const int j  = threadIdx.x;   // 0..63
    const int yr = threadIdx.y;   // 0..3
    const long ri = (long)blockIdx.x * 4 + yr;

    int t, hh, base, mode;
    if (ri < (long)T_ * H_) {
        mode = 0; t = (int)(ri / H_); hh = (int)(ri % H_); base = hh * D_;
    } else if (ri < (long)T_ * (H_ + HKV_)) {
        long r2 = ri - (long)T_ * H_;
        mode = 1; t = (int)(r2 / HKV_); hh = (int)(r2 % HKV_); base = H_ * D_ + hh * D_;
    } else {
        long r2 = ri - (long)T_ * (H_ + HKV_);
        mode = 2; t = (int)(r2 / HKV_); hh = (int)(r2 % HKV_); base = (H_ + HKV_) * D_ + hh * D_;
    }

    const float* src = qkv + (size_t)t * QKVW_ + base;
    float x1 = src[j];
    float x2 = src[j + 64];

    // sum of squares over the 128-elem row (2 waves per row)
    float p = x1 * x1 + x2 * x2;
    p += __shfl_xor(p, 1);  p += __shfl_xor(p, 2);
    p += __shfl_xor(p, 4);  p += __shfl_xor(p, 8);
    p += __shfl_xor(p, 16);
    if ((j & 31) == 0) red[yr][j >> 5] = p;
    __syncthreads();
    float ss = red[yr][0] + red[yr][1];

    if (mode < 2) {
        float inv = rsqrtf(ss * (1.0f / 128.0f) + 1e-6f);
        const float* wgt = (mode == 0) ? qw : kw;
        x1 = x1 * inv * wgt[j];
        x2 = x2 * inv * wgt[j + 64];
        float pf   = (float)pos[t];
        float invf = __expf(-((float)(2 * j) * (1.0f / 128.0f)) * 9.210340371976184f); // ln(1e4)
        float fr = pf * invf, s, c;
        __sincosf(fr, &s, &c);
        float y1 = x1 * c - x2 * s;
        float y2 = x2 * c + x1 * s;
        __bf16* dst = ((mode == 0) ? qB : kB) + ((size_t)hh * T_ + t) * D_;
        dst[j]      = f2bf(y1);
        dst[j + 64] = f2bf(y2);
    } else {
        __bf16* dst = vT + (size_t)hh * D_ * T_;
        dst[(size_t)j * T_ + t]        = f2bf(x1);
        dst[(size_t)(j + 64) * T_ + t] = f2bf(x2);
    }
}

// ---------------------------------------------------------------------------
// Kernel 3/4: causal flash attention, bf16 WMMA, fp32 accum.
// Grid (T/64, H); block 128 thr (4 waves, 16 query rows each).
// K/V tiles staged with GLOBAL_LOAD_ASYNC_TO_LDS_B128 (no VGPR round-trip).
// ---------------------------------------------------------------------------
#define KTSTR 136   // Kt row stride (272B, 16B aligned)
#define VTSTR 48

__global__ __launch_bounds__(128)
void flash_attn(const __bf16* __restrict__ qB, const __bf16* __restrict__ kB,
                const __bf16* __restrict__ vT, __bf16* __restrict__ attnB) {
    __shared__ __bf16 Kt[32 * KTSTR];     // [key][d]  (== transposed-B layout)
    __shared__ __bf16 Vt[128 * VTSTR];    // [d][key]  (== transposed-B layout)
    __shared__ __bf16 Pl[4][16 * 32];     // per-wave P tile, A layout

    const int tid  = threadIdx.x;
    const int lane = tid & 31;
    const int w    = tid >> 5;
    const int qb   = blockIdx.x;
    const int h    = blockIdx.y;
    const int kv   = h >> 3;              // GQA: 8 q-heads per kv-head
    const int m0   = qb * 64 + w * 16;
    const int g    = lane >> 4;
    const int nlo  = lane & 15;

    // Q fragments resident in registers (16 rows x 128 D)
    bf16x16 qf[4];
    {
        const __bf16* qbase = qB + ((size_t)h * T_ + m0) * D_;
#pragma unroll
        for (int kk = 0; kk < 4; ++kk)
            qf[kk] = ldfragA(qbase, D_, kk * 32, lane);
    }

    f32x8 o[8];
#pragma unroll
    for (int ni = 0; ni < 8; ++ni)
#pragma unroll
        for (int i = 0; i < 8; ++i) o[ni][i] = 0.0f;
    float ms[8], ls[8];
#pragma unroll
    for (int i = 0; i < 8; ++i) { ms[i] = -1e30f; ls[i] = 0.0f; }

    const float scale = 0.08838834764831845f;  // 1/sqrt(128)
    const int nkt = qb * 2 + 2;                // causal bound: keys <= qb*64+63

    for (int kt = 0; kt < nkt; ++kt) {
        // ---- async-stage 32-key K and V^T tiles straight into LDS ----
        const __bf16* ksrc = kB + (size_t)kv * T_ * D_ + (size_t)kt * 32 * D_;
#pragma unroll
        for (int it = 0; it < 4; ++it) {
            int slot = tid + it * 128;
            int r = slot >> 4;            // key 0..31
            int c = (slot & 15) << 3;     // d 0..120
            async_g2l_b128(ksrc + (size_t)r * D_ + c, &Kt[r * KTSTR + c]);
        }
        const __bf16* vsrc = vT + (size_t)kv * D_ * T_ + (size_t)kt * 32;
#pragma unroll
        for (int it = 0; it < 4; ++it) {
            int slot = tid + it * 128;
            int r = slot >> 2;            // d 0..127
            int c = (slot & 3) << 3;      // key 0..24
            async_g2l_b128(vsrc + (size_t)r * T_ + c, &Vt[r * VTSTR + c]);
        }
        asm volatile("s_wait_asynccnt 0x0" ::: "memory");  // own copies landed
        __syncthreads();                                   // all waves' copies

        // ---- S = Q K^T (16x32), K-dim = D = 4 x 32 ----
        f32x8 s0, s1;
#pragma unroll
        for (int i = 0; i < 8; ++i) { s0[i] = 0.0f; s1[i] = 0.0f; }
#pragma unroll
        for (int kk = 0; kk < 4; ++kk) {
            bf16x16 b0 = ldfragB(&Kt[0],           KTSTR, kk * 32, lane);
            bf16x16 b1 = ldfragB(&Kt[16 * KTSTR],  KTSTR, kk * 32, lane);
            s0 = wmma_bf16(qf[kk], b0, s0);
            s1 = wmma_bf16(qf[kk], b1, s1);
        }

        // ---- causal mask + online softmax (16-lane groups own full rows) ----
        float alpha[8];
#pragma unroll
        for (int i = 0; i < 8; ++i) {
            int row = m0 + i + 8 * g;
            int c0  = kt * 32 + nlo;
            float a = s0[i] * scale; if (c0      > row) a = -1e30f;
            float b = s1[i] * scale; if (c0 + 16 > row) b = -1e30f;
            float rm = fmaxf(a, b);
            rm = fmaxf(rm, __shfl_xor(rm, 1));
            rm = fmaxf(rm, __shfl_xor(rm, 2));
            rm = fmaxf(rm, __shfl_xor(rm, 4));
            rm = fmaxf(rm, __shfl_xor(rm, 8));
            float mn = fmaxf(ms[i], rm);
            float al = __expf(ms[i] - mn);
            float p0 = __expf(a - mn);
            float p1 = __expf(b - mn);
            s0[i] = p0; s1[i] = p1;
            float rs = p0 + p1;
            rs += __shfl_xor(rs, 1); rs += __shfl_xor(rs, 2);
            rs += __shfl_xor(rs, 4); rs += __shfl_xor(rs, 8);
            ls[i] = ls[i] * al + rs;
            ms[i] = mn;
            alpha[i] = al;
        }
#pragma unroll
        for (int ni = 0; ni < 8; ++ni)
#pragma unroll
            for (int i = 0; i < 8; ++i) o[ni][i] = o[ni][i] * alpha[i];

        // ---- round-trip P through per-wave LDS into A-fragment layout ----
        __bf16* pw = &Pl[w][0];
#pragma unroll
        for (int i = 0; i < 8; ++i) {
            int row = i + 8 * g;
            pw[row * 32 + nlo]      = f2bf(s0[i]);
            pw[row * 32 + nlo + 16] = f2bf(s1[i]);
        }
        asm volatile("s_wait_dscnt 0x0" ::: "memory");   // own-wave LDS RAW
        bf16x16 pf = ldfragA(pw, 32, 0, lane);

        // ---- O += P V  (K-dim = 32 keys, one WMMA per 16-wide D slice) ----
#pragma unroll
        for (int ni = 0; ni < 8; ++ni) {
            bf16x16 vf = ldfragB(&Vt[ni * 16 * VTSTR], VTSTR, 0, lane);
            o[ni] = wmma_bf16(pf, vf, o[ni]);
        }
        __syncthreads();
    }

    // ---- normalize and emit bf16 attn rows [T][H*D] ----
#pragma unroll
    for (int ni = 0; ni < 8; ++ni)
#pragma unroll
        for (int i = 0; i < 8; ++i) {
            int row = m0 + i + 8 * g;
            int col = h * D_ + ni * 16 + nlo;
            attnB[(size_t)row * NOUT_ + col] = f2bf(o[ni][i] / ls[i]);
        }
}

// ---------------------------------------------------------------------------
extern "C" void kernel_launch(void* const* d_in, const int* in_sizes, int n_in,
                              void* d_out, int out_size, void* d_ws, size_t ws_size,
                              hipStream_t stream) {
    const float* hidden = (const float*)d_in[0];
    const int*   pos    = (const int*)  d_in[1];
    const float* w_qkv  = (const float*)d_in[2];
    const float* w_o    = (const float*)d_in[3];
    const float* qw     = (const float*)d_in[4];
    const float* kw     = (const float*)d_in[5];
    float*       out    = (float*)d_out;

    char* ws = (char*)d_ws;
    size_t off = 0;
    auto carve = [&](size_t bytes) {
        char* p = ws + off;
        off = (off + bytes + 255) & ~(size_t)255;
        return p;
    };
    float*  qkv   = (float*) carve((size_t)T_ * QKVW_ * sizeof(float));     // 40 MB
    __bf16* qB    = (__bf16*)carve((size_t)H_ * T_ * D_ * 2);               // 16 MB
    __bf16* kB    = (__bf16*)carve((size_t)HKV_ * T_ * D_ * 2);             //  2 MB
    __bf16* vT    = (__bf16*)carve((size_t)HKV_ * D_ * T_ * 2);             //  2 MB
    __bf16* attnB = (__bf16*)carve((size_t)T_ * NOUT_ * 2);                 // 16 MB

    // 1) QKV projection: (2048x2048) x (2048x5120)
    gemm_wmma<false><<<dim3(QKVW_ / BN, T_ / BM), 256, 0, stream>>>(
        hidden, w_qkv, qkv, T_, QKVW_, HID_);

    // 2) RMSNorm + RoPE + layout conversion
    {
        int rows = T_ * (H_ + 2 * HKV_);          // 81920
        qkv_post<<<rows / 4, dim3(64, 4), 0, stream>>>(qkv, pos, qw, kw, qB, kB, vT);
    }

    // 3) causal flash attention
    flash_attn<<<dim3(T_ / 64, H_), 128, 0, stream>>>(qB, kB, vT, attnB);

    // 4) output projection: (2048x4096) x (4096x2048)
    gemm_wmma<true><<<dim3(HID_ / BN, T_ / BM), 256, 0, stream>>>(
        attnB, w_o, out, T_, HID_, NOUT_);
}
